// Qwen2FP8MLP_89111981457456
// MI455X (gfx1250) — compile-verified, matrix operand order checked by reading, not run on hardware
//
#include <hip/hip_runtime.h>

// ---------------------------------------------------------------------------
// Qwen2 FP8 MLP (w8a16): x[T,H] @ dq(w_gate_up)[2I,H]^T -> silu*mul -> h[T,I]
//                        h[T,I] @ dq(w_down)[H,I]^T -> out[T,H]
// fp8-e4m3 values are exact in f16 -> dequant weights to f16, activations f16,
// accumulate f32 via V_WMMA_F32_16X16X32_F16 (wave32, CDNA5).
// Block tile M=64 x N=128 (+128 up), 8 waves (2x4), wave tile 32x32,
// software-pipelined global->reg->LDS staging.
// ---------------------------------------------------------------------------

#define HIDDEN 3584
#define INTER  18944
#define TOKENS 4096

typedef __attribute__((ext_vector_type(16))) _Float16 v16h;
typedef __attribute__((ext_vector_type(8)))  _Float16 v8h;
typedef __attribute__((ext_vector_type(4)))  _Float16 v4h;
typedef __attribute__((ext_vector_type(8)))  float    v8f;
typedef __attribute__((ext_vector_type(4)))  float    v4f;

union Frag { v16h v; v8h h[2]; };

#define LDS_PITCH 40   // halves per row: 80B, multiple of 16B, bank-conflict-free

#define WMMA_F16(A, B, C) \
    __builtin_amdgcn_wmma_f32_16x16x32_f16(false, (A), false, (B), (short)0, (C), false, false)

// -------------------------- kernel 0: x f32 -> f16 -------------------------
__global__ void __launch_bounds__(256)
cvt_f32_to_f16(const float* __restrict__ src, _Float16* __restrict__ dst, int n) {
    int i = (blockIdx.x * 256 + threadIdx.x) * 4;
    if (i + 3 < n) {
        v4f f = *(const v4f*)(src + i);
        v4h h = { (_Float16)f[0], (_Float16)f[1], (_Float16)f[2], (_Float16)f[3] };
        *(v4h*)(dst + i) = h;
    }
}

// Load one 16x32 f16 A/B fragment from LDS (row-major, pitch LDS_PITCH).
// Per ISA 16-bit 16x32 layout: lanes 0-15 hold K=0..7 (v0-3) & 16..23 (v4-7);
// lanes 16-31 hold K=8..15 & 24..31.  Two ds_load_b128 per fragment.
__device__ inline v16h load_frag(const _Float16* lds, int row, int lane) {
    const int hsel = lane >> 4;
    Frag f;
    f.h[0] = *(const v8h*)(lds + row * LDS_PITCH + hsel * 8);
    f.h[1] = *(const v8h*)(lds + row * LDS_PITCH + 16 + hsel * 8);
    return f.v;
}

__device__ inline v4h dq4(v4f w, float s) {
    v4h r = { (_Float16)(w[0] * s), (_Float16)(w[1] * s),
              (_Float16)(w[2] * s), (_Float16)(w[3] * s) };
    return r;
}

__device__ inline float silu_mul(float g, float u) {
    // silu(g)*u with fast v_rcp_f32 (no IEEE div expansion)
    return g * __builtin_amdgcn_rcpf(1.0f + __expf(-g)) * u;
}

// ------------------- kernel 1: gate_up GEMM + SiLU * mul -------------------
// grid: (TOKENS/64, INTER/128).
__global__ void __launch_bounds__(256)
gemm_gate_up_silu(const _Float16* __restrict__ xh,   // [T, H] f16
                  const float*    __restrict__ w,    // [2I, H] f32 (fp8 values)
                  const float*    __restrict__ s,    // [2I] scales
                  _Float16*       __restrict__ hout) // [T, I] f16
{
    __shared__ _Float16 ldsA [64  * LDS_PITCH];
    __shared__ _Float16 ldsBg[128 * LDS_PITCH];
    __shared__ _Float16 ldsBu[128 * LDS_PITCH];

    const int tid  = threadIdx.x;
    const int lane = tid & 31;
    const int wave = tid >> 5;
    const int wm   = wave >> 2;        // 0..1  (M sub-tile, 32 rows each)
    const int wn   = wave & 3;         // 0..3  (N sub-tile, 32 cols each)
    const int rowBase = blockIdx.x * 64;
    const int colBase = blockIdx.y * 128;

    // ---- loop-invariant staging coordinates ----
    const int aRow = tid >> 2;               // 0..63
    const int aCol = (tid & 3) * 8;          // 0,8,16,24  (v8h = 16B)
    const _Float16* aSrc = xh + (size_t)(rowBase + aRow) * HIDDEN + aCol;

    const int bCol  = (tid & 7) * 4;         // 0..28      (v4f = 16B)
    const int bRow0 = tid >> 3;              // 0..31, +it*32
    const float* bgSrc[4]; const float* buSrc[4];
    float sg[4], su[4];
    #pragma unroll
    for (int it = 0; it < 4; ++it) {
        int row = bRow0 + it * 32;           // 0..127
        bgSrc[it] = w + (size_t)(colBase + row) * HIDDEN + bCol;
        buSrc[it] = w + (size_t)(INTER + colBase + row) * HIDDEN + bCol;
        sg[it] = s[colBase + row];
        su[it] = s[INTER + colBase + row];
    }

    v8f cg[2][2] = {{{}, {}}, {{}, {}}};
    v8f cu[2][2] = {{{}, {}}, {{}, {}}};

    // ---- prologue: prefetch K-tile 0 into registers ----
    v8h aReg; v4f wgReg[4], wuReg[4];
    aReg = *(const v8h*)(aSrc);
    #pragma unroll
    for (int it = 0; it < 4; ++it) {
        wgReg[it] = *(const v4f*)(bgSrc[it]);
        wuReg[it] = *(const v4f*)(buSrc[it]);
    }

    for (int kk = 0; kk < HIDDEN; kk += 32) {
        // ---- regs -> LDS (dequant weights on the way) ----
        *(v8h*)(&ldsA[aRow * LDS_PITCH + aCol]) = aReg;
        #pragma unroll
        for (int it = 0; it < 4; ++it) {
            int row = bRow0 + it * 32;
            *(v4h*)(&ldsBg[row * LDS_PITCH + bCol]) = dq4(wgReg[it], sg[it]);
            *(v4h*)(&ldsBu[row * LDS_PITCH + bCol]) = dq4(wuReg[it], su[it]);
        }
        __syncthreads();

        // ---- prefetch next K-tile while WMMAs run ----
        const int kn = kk + 32;
        if (kn < HIDDEN) {
            aReg = *(const v8h*)(aSrc + kn);
            #pragma unroll
            for (int it = 0; it < 4; ++it) {
                wgReg[it] = *(const v4f*)(bgSrc[it] + kn);
                wuReg[it] = *(const v4f*)(buSrc[it] + kn);
            }
        }

        // ---- compute: 8 WMMAs per wave ----
        const int m = lane & 15;
        v16h a0  = load_frag(ldsA,  wm * 32 + m,       lane);
        v16h a1  = load_frag(ldsA,  wm * 32 + 16 + m,  lane);
        v16h bg0 = load_frag(ldsBg, wn * 32 + m,       lane);
        v16h bg1 = load_frag(ldsBg, wn * 32 + 16 + m,  lane);
        v16h bu0 = load_frag(ldsBu, wn * 32 + m,       lane);
        v16h bu1 = load_frag(ldsBu, wn * 32 + 16 + m,  lane);

        cg[0][0] = WMMA_F16(a0, bg0, cg[0][0]);
        cg[0][1] = WMMA_F16(a0, bg1, cg[0][1]);
        cg[1][0] = WMMA_F16(a1, bg0, cg[1][0]);
        cg[1][1] = WMMA_F16(a1, bg1, cg[1][1]);
        cu[0][0] = WMMA_F16(a0, bu0, cu[0][0]);
        cu[0][1] = WMMA_F16(a0, bu1, cu[0][1]);
        cu[1][0] = WMMA_F16(a1, bu0, cu[1][0]);
        cu[1][1] = WMMA_F16(a1, bu1, cu[1][1]);
        __syncthreads();
    }

    // ---- epilogue: silu(gate)*up -> h (f16), fully unrolled ----
    // C layout: VGPR r, lanes 0-15 -> M=r, N=lane; lanes 16-31 -> M=r+8, N=lane-16.
    const int mBase = rowBase + wm * 32 + (lane >> 4) * 8;
    const int nBase = colBase + wn * 32 + (lane & 15);
    #pragma unroll
    for (int i = 0; i < 2; ++i) {
        #pragma unroll
        for (int r = 0; r < 8; ++r) {
            const size_t rowOff = (size_t)(mBase + i * 16 + r) * INTER;
            #pragma unroll
            for (int j = 0; j < 2; ++j) {
                float hv = silu_mul(cg[i][j][r], cu[i][j][r]);
                hout[rowOff + nBase + j * 16] = (_Float16)hv;
            }
        }
    }
}

// --------------------------- kernel 2: down GEMM ---------------------------
// grid: (TOKENS/64, HIDDEN/128).  out[T,H] f32.
__global__ void __launch_bounds__(256)
gemm_down(const _Float16* __restrict__ hin,  // [T, I] f16
          const float*    __restrict__ w,    // [H, I] f32 (fp8 values)
          const float*    __restrict__ s,    // [H] scales
          float*          __restrict__ out)  // [T, H] f32
{
    __shared__ _Float16 ldsA[64  * LDS_PITCH];
    __shared__ _Float16 ldsB[128 * LDS_PITCH];

    const int tid  = threadIdx.x;
    const int lane = tid & 31;
    const int wave = tid >> 5;
    const int wm   = wave >> 2;
    const int wn   = wave & 3;
    const int rowBase = blockIdx.x * 64;
    const int colBase = blockIdx.y * 128;

    const int aRow = tid >> 2;
    const int aCol = (tid & 3) * 8;
    const _Float16* aSrc = hin + (size_t)(rowBase + aRow) * INTER + aCol;

    const int bCol  = (tid & 7) * 4;
    const int bRow0 = tid >> 3;
    const float* bSrc[4]; float sc[4];
    #pragma unroll
    for (int it = 0; it < 4; ++it) {
        int row = bRow0 + it * 32;
        bSrc[it] = w + (size_t)(colBase + row) * INTER + bCol;
        sc[it] = s[colBase + row];
    }

    v8f c[2][2] = {{{}, {}}, {{}, {}}};

    v8h aReg; v4f wReg[4];
    aReg = *(const v8h*)(aSrc);
    #pragma unroll
    for (int it = 0; it < 4; ++it) wReg[it] = *(const v4f*)(bSrc[it]);

    for (int kk = 0; kk < INTER; kk += 32) {
        *(v8h*)(&ldsA[aRow * LDS_PITCH + aCol]) = aReg;
        #pragma unroll
        for (int it = 0; it < 4; ++it) {
            int row = bRow0 + it * 32;
            *(v4h*)(&ldsB[row * LDS_PITCH + bCol]) = dq4(wReg[it], sc[it]);
        }
        __syncthreads();

        const int kn = kk + 32;
        if (kn < INTER) {
            aReg = *(const v8h*)(aSrc + kn);
            #pragma unroll
            for (int it = 0; it < 4; ++it) wReg[it] = *(const v4f*)(bSrc[it] + kn);
        }

        const int m = lane & 15;
        v16h a0 = load_frag(ldsA, wm * 32 + m,      lane);
        v16h a1 = load_frag(ldsA, wm * 32 + 16 + m, lane);
        v16h b0 = load_frag(ldsB, wn * 32 + m,      lane);
        v16h b1 = load_frag(ldsB, wn * 32 + 16 + m, lane);

        c[0][0] = WMMA_F16(a0, b0, c[0][0]);
        c[0][1] = WMMA_F16(a0, b1, c[0][1]);
        c[1][0] = WMMA_F16(a1, b0, c[1][0]);
        c[1][1] = WMMA_F16(a1, b1, c[1][1]);
        __syncthreads();
    }

    const int mBase = rowBase + wm * 32 + (lane >> 4) * 8;
    const int nBase = colBase + wn * 32 + (lane & 15);
    #pragma unroll
    for (int i = 0; i < 2; ++i) {
        #pragma unroll
        for (int r = 0; r < 8; ++r) {
            const size_t rowOff = (size_t)(mBase + i * 16 + r) * HIDDEN;
            #pragma unroll
            for (int j = 0; j < 2; ++j)
                out[rowOff + nBase + j * 16] = c[i][j][r];
        }
    }
}

// ------------------------------- launcher ----------------------------------
extern "C" void kernel_launch(void* const* d_in, const int* in_sizes, int n_in,
                              void* d_out, int out_size, void* d_ws, size_t ws_size,
                              hipStream_t stream) {
    const float* x    = (const float*)d_in[0];  // [T, H]
    const float* w_gu = (const float*)d_in[1];  // [2I, H]
    const float* s_gu = (const float*)d_in[2];  // [2I]
    const float* w_d  = (const float*)d_in[3];  // [H, I]
    const float* s_d  = (const float*)d_in[4];  // [H]
    float* out = (float*)d_out;

    _Float16* xh   = (_Float16*)d_ws;                       // T*H f16
    _Float16* hbuf = xh + (size_t)TOKENS * HIDDEN;          // T*I f16

    const int nx = TOKENS * HIDDEN;
    cvt_f32_to_f16<<<dim3(nx / 1024), 256, 0, stream>>>(x, xh, nx);

    gemm_gate_up_silu<<<dim3(TOKENS / 64, INTER / 128), 256, 0, stream>>>(
        xh, w_gu, s_gu, hbuf);

    gemm_down<<<dim3(TOKENS / 64, HIDDEN / 128), 256, 0, stream>>>(
        hbuf, w_d, s_d, out);

    (void)in_sizes; (void)n_in; (void)out_size; (void)ws_size;
}
